// Block_9122510537233
// MI455X (gfx1250) — matrix operand order repaired and checked by reading, not compiled, not run on hardware
//
#include <hip/hip_runtime.h>
#include <hip/hip_bf16.h>
#include <math.h>

#define B_   16
#define N_   8192
#define C_   256
#define H_   8
#define DH_  32
#define NHASH 4
#define NBUCK 8
#define M_   32
#define DFF_ 1024

typedef __attribute__((ext_vector_type(16))) _Float16 v16h;
typedef __attribute__((ext_vector_type(8)))  _Float16 v8h;
typedef __attribute__((ext_vector_type(8)))  float    v8f;
typedef __attribute__((ext_vector_type(4)))  float    v4f;
typedef __attribute__((ext_vector_type(4)))  unsigned int v4u;
typedef __attribute__((ext_vector_type(8)))  int      v8i;
typedef __attribute__((ext_vector_type(4)))  int      v4i;

__device__ __forceinline__ v8f wmma16(v16h a, v16h b, v8f c) {
  // D = A(16x32 f16) * B(32x16 f16) + C(16x16 f32)
  return __builtin_amdgcn_wmma_f32_16x16x32_f16(false, a, false, b, (short)0, c, false, false);
}

// A-fragment from LDS (row-major f16 tile, padded stride).
// ISA 16-bit A 16x32 layout: lanes0-15 row=lane K={0..7,16..23}; lanes16-31 K={8..15,24..31}
__device__ __forceinline__ v16h load_a_lds(const _Float16* lds, int row_base, int lda,
                                           int k_base, int lane) {
  int r = row_base + (lane & 15);
  const _Float16* p = lds + r * lda + k_base + ((lane >> 4) << 3);
  v16h a;
#pragma unroll
  for (int j = 0; j < 8; ++j) { a[j] = p[j]; a[8 + j] = p[16 + j]; }
  return a;
}

// B-fragment from pre-swizzled weights: each lane's 16 halves contiguous (32B)
__device__ __forceinline__ v16h load_b_swz(const _Float16* w, int tile, int lane) {
  return *reinterpret_cast<const v16h*>(w + ((size_t)tile * 32 + lane) * 16);
}

// C/D (16x16 f32): VGPR r -> row r (lanes0-15) / r+8 (lanes16-31), col = lane&15
__device__ __forceinline__ void store_c_f16_lds(_Float16* lds, int row_base, int lda,
                                                int col_base, v8f acc, int lane) {
  int col = col_base + (lane & 15);
  int ro  = (lane >> 4) << 3;
#pragma unroll
  for (int r = 0; r < 8; ++r) lds[(row_base + ro + r) * lda + col] = (_Float16)acc[r];
}

// TDM: DMA a 2D f16 tile (rows x 256 halves, global row stride 256) into LDS with
// hardware padding of 4 DWORDs after every 128 DWORDs (-> 264-half LDS row stride).
__device__ __forceinline__ void tdm_load_tile_264(const _Float16* gsrc, _Float16* ldst,
                                                  int rows) {
  unsigned long long ga = (unsigned long long)(uintptr_t)gsrc;
  unsigned int lds_addr = (unsigned int)(uintptr_t)ldst;  // aperture low 32 = LDS byte addr
  v4u g0;
  g0[0] = 1u;                                  // count=1, user descriptor
  g0[1] = lds_addr;                            // lds_addr
  g0[2] = (unsigned int)(ga & 0xffffffffu);    // global_addr[31:0]
  g0[3] = (unsigned int)((ga >> 32) & 0x1ffffffu) | (2u << 30);  // addr[56:32] | type=2
  v8i g1;
  // data_size=2B(code1) | pad_enable | pad_interval=6(128 DW) | pad_amount=3(4 DW)
  g1[0] = (int)((1u << 16) | (1u << 20) | (6u << 22) | (3u << 25));
  g1[1] = (int)(256u << 16);                   // tensor_dim0=256 (bits 79:48, low half)
  g1[2] = (int)(((unsigned)N_ & 0xffffu) << 16); // tensor_dim0 hi=0 | tensor_dim1 lo
  g1[3] = (int)(256u << 16);                   // tensor_dim1 hi=0 | tile_dim0=256
  g1[4] = rows;                                // tile_dim1=rows, tile_dim2=0
  g1[5] = 256;                                 // tensor_dim0_stride low32 = 256
  g1[6] = 0;                                   // stride hi | tensor_dim1_stride lo
  g1[7] = 0;
  v4i gz; gz[0] = 0; gz[1] = 0; gz[2] = 0; gz[3] = 0;
  v8i gz8; gz8[0] = 0; gz8[1] = 0; gz8[2] = 0; gz8[3] = 0;
  gz8[4] = 0; gz8[5] = 0; gz8[6] = 0; gz8[7] = 0;
  __builtin_amdgcn_tensor_load_to_lds(g0, g1, gz, gz, gz8, 0);
}

__global__ void zero_kernel(float* p, int n) {
  int i = blockIdx.x * blockDim.x + threadIdx.x;
  if (i < n) p[i] = 0.f;
}

// Swizzle f32 weight [K][NC] row-major into per-tile WMMA-B layout (f16).
// tile t = kt*(NC/16)+ct ; lane: col = ct*16+(lane&15) ; elem h: k = kt*32 + (h&7) + (lane>=16)*8 + (h>=8)*16
__global__ void swz_kernel(const float* __restrict__ w, _Float16* __restrict__ dst,
                           int K, int NC) {
  int idx = blockIdx.x * blockDim.x + threadIdx.x;
  if (idx >= K * NC) return;
  int h    = idx & 15;
  int lane = (idx >> 4) & 31;
  int t    = idx >> 9;
  int ctiles = NC >> 4;
  int kt = t / ctiles, ct = t % ctiles;
  int col = ct * 16 + (lane & 15);
  int k   = kt * 32 + (h & 7) + ((lane >> 4) << 3) + ((h >> 3) << 4);
  dst[idx] = (_Float16)w[(size_t)k * NC + col];
}

// LN1 + write x_hat (f16) + LSH bucket assignment + pooled scatter (LDS then global atomics)
__global__ void ln_hash_kernel(const float* __restrict__ x, const float* __restrict__ rot,
                               const float* __restrict__ g, const float* __restrict__ bb,
                               _Float16* __restrict__ xh, float* __restrict__ pooled_g,
                               float* __restrict__ counts_g) {
  extern __shared__ char smem[];
  float* rotl = (float*)smem;          // 256*16
  float* pool = rotl + C_ * 16;        // 32*256
  float* cnt  = pool + 32 * C_;        // 32
  float* gl   = cnt + 32;              // 256
  float* bl   = gl + C_;               // 256
  int tid = threadIdx.x, lane = tid & 31, w = tid >> 5;
  int b = blockIdx.y;
  int base = blockIdx.x * 256;

  for (int i = tid; i < C_ * 16; i += 256) rotl[i] = rot[i];
  for (int i = tid; i < 32 * C_; i += 256) pool[i] = 0.f;
  if (tid < 32) cnt[tid] = 0.f;
  if (tid < C_) { gl[tid] = g[tid]; bl[tid] = bb[tid]; }
  __syncthreads();

  for (int i = 0; i < 32; ++i) {
    int n = base + w * 32 + i;
    const float* xr = x + ((size_t)b * N_ + n) * C_ + lane * 8;
    v4f a0 = *(const v4f*)xr;
    v4f a1 = *(const v4f*)(xr + 4);
    float v[8] = {a0[0], a0[1], a0[2], a0[3], a1[0], a1[1], a1[2], a1[3]};
    float s = 0.f, ss = 0.f;
#pragma unroll
    for (int j = 0; j < 8; ++j) { s += v[j]; ss += v[j] * v[j]; }
#pragma unroll
    for (int off = 16; off >= 1; off >>= 1) {
      s += __shfl_xor(s, off, 32);
      ss += __shfl_xor(ss, off, 32);
    }
    float mean = s * (1.f / C_);
    float var  = ss * (1.f / C_) - mean * mean;
    float rstd = rsqrtf(var + 1e-5f);
    float xn[8];
    v8h hv;
#pragma unroll
    for (int j = 0; j < 8; ++j) {
      xn[j] = (v[j] - mean) * rstd * gl[lane * 8 + j] + bl[lane * 8 + j];
      hv[j] = (_Float16)xn[j];
    }
    *(v8h*)(xh + ((size_t)b * N_ + n) * C_ + lane * 8) = hv;

    // rotated = xn @ rot  (16 outputs)
    float rl[16];
#pragma unroll
    for (int q = 0; q < 16; ++q) rl[q] = 0.f;
#pragma unroll
    for (int j = 0; j < 8; ++j) {
      const float* rr = rotl + (lane * 8 + j) * 16;
      float xv = xn[j];
#pragma unroll
      for (int q = 0; q < 16; ++q) rl[q] += xv * rr[q];
    }
#pragma unroll
    for (int q = 0; q < 16; ++q) {
#pragma unroll
      for (int off = 16; off >= 1; off >>= 1) rl[q] += __shfl_xor(rl[q], off, 32);
    }
    // argmax over [r, -r] per hash (first-max tie-break)
#pragma unroll
    for (int hsh = 0; hsh < NHASH; ++hsh) {
      float best = rl[hsh * 4]; int bi = 0;
#pragma unroll
      for (int q = 1; q < 4; ++q) if (rl[hsh * 4 + q] > best) { best = rl[hsh * 4 + q]; bi = q; }
#pragma unroll
      for (int q = 0; q < 4; ++q) { float nv = -rl[hsh * 4 + q]; if (nv > best) { best = nv; bi = 4 + q; } }
      int slot = hsh * NBUCK + bi;
#pragma unroll
      for (int j = 0; j < 8; ++j) atomicAdd(&pool[slot * C_ + lane * 8 + j], xn[j]);
      if (lane == 0) atomicAdd(&cnt[slot], 1.f);
    }
  }
  __syncthreads();
  for (int i = tid; i < 32 * C_; i += 256)
    atomicAdd(&pooled_g[(size_t)b * 32 * C_ + i], pool[i]);
  if (tid < 32) atomicAdd(&counts_g[b * 32 + tid], cnt[tid]);
}

// finalize pooled -> rp_x, project to K,V (f16, per-head layout). Tiny GEMM: VALU.
__global__ void kv_kernel(const float* __restrict__ pooled, const float* __restrict__ counts,
                          const float* __restrict__ kv_w, _Float16* __restrict__ kh,
                          _Float16* __restrict__ vh) {
  extern __shared__ char smem[];
  float* rp = (float*)smem;  // 32*256
  int tid = threadIdx.x, b = blockIdx.x;
  for (int i = tid; i < 32 * C_; i += 256) {
    int m = i >> 8;
    rp[i] = pooled[(size_t)b * 32 * C_ + i] / (counts[b * 32 + m] + 1e-20f);
  }
  __syncthreads();
  for (int e = tid; e < M_ * 2 * C_; e += 256) {
    int m = e >> 9;
    int col = e & 511;
    const float* row = rp + m * C_;
    float acc = 0.f;
    for (int f = 0; f < C_; ++f) acc += row[f] * kv_w[(size_t)f * (2 * C_) + col];
    int s = col >> 8;       // kv reshape (M,2,H,Dh): 0 = k, 1 = v
    int c = col & 255;
    int head = c >> 5, d = c & 31;
    _Float16* dst = s ? vh : kh;
    dst[(((size_t)b * H_ + head) * M_ + m) * DH_ + d] = (_Float16)acc;
  }
}

// Q = x_hat @ q_w (WMMA, x_hat tile staged by TDM), masked softmax attention, +x residual
__global__ void q_attn_kernel(const float* __restrict__ x, const _Float16* __restrict__ xh,
                              const _Float16* __restrict__ qswz, const _Float16* __restrict__ kh,
                              const _Float16* __restrict__ vh, const float* __restrict__ counts,
                              float* __restrict__ out) {
  extern __shared__ char smem[];
  float* maskl  = (float*)smem;                  // 32 (+pad to 128B)
  _Float16* xt  = (_Float16*)(smem + 128);       // 64 x 264
  _Float16* khl = xt + 64 * 264;                 // 8*32*32
  _Float16* vhl = khl + 8192;                    // 8*32*32
  _Float16* qh  = vhl + 8192;                    // 64 x 264
  int tid = threadIdx.x, lane = tid & 31, w = tid >> 5;
  int b  = blockIdx.y;
  int n0 = blockIdx.x * 64;

  // async DMA x_hat tile -> LDS (one wave issues; TENSORcnt waited before barrier)
  if (w == 0) {
    tdm_load_tile_264(xh + ((size_t)b * N_ + n0) * C_, xt, 64);
    __builtin_amdgcn_s_wait_tensorcnt(0);
  }
  for (int i = tid; i < 1024; i += 256) {
    *(v8h*)(khl + i * 8) = *(const v8h*)(kh + (size_t)b * 8192 + i * 8);
    *(v8h*)(vhl + i * 8) = *(const v8h*)(vh + (size_t)b * 8192 + i * 8);
  }
  if (tid < 32) maskl[tid] = (counts[b * 32 + tid] >= 1.f) ? 1.f : 0.f;
  __syncthreads();

  // Q projection: wave -> (row-tile rt, column half ch); A frags hoisted to registers
  int rt = w & 3, ch = w >> 2;
  const float scale = 0.17677669529663687f;  // Dh^-0.5
  v16h afrag[8];
#pragma unroll
  for (int kt = 0; kt < 8; ++kt) afrag[kt] = load_a_lds(xt, rt * 16, 264, kt * 32, lane);
  for (int c8 = 0; c8 < 8; ++c8) {
    int ct = ch * 8 + c8;
    v8f acc = {};
#pragma unroll
    for (int kt = 0; kt < 8; ++kt) {
      acc = wmma16(afrag[kt], load_b_swz(qswz, kt * 16 + ct, lane), acc);
    }
#pragma unroll
    for (int r = 0; r < 8; ++r) acc[r] *= scale;
    store_c_f16_lds(qh, rt * 16, 264, ct * 16, acc, lane);
  }
  __syncthreads();

  // attention: 64 rows x 8 heads = 512 tasks, 2 per thread (M=32 -> VALU is cheap)
  for (int t2 = 0; t2 < 2; ++t2) {
    int task = tid + t2 * 256;
    int row = task >> 3, head = task & 7;
    const _Float16* qr = qh + row * 264 + head * 32;
    float p[32];
    float mx = -3.402823466e38f;
    for (int m = 0; m < 32; ++m) {
      const _Float16* kr = khl + (head * 32 + m) * 32;
      float acc = 0.f;
#pragma unroll
      for (int d = 0; d < 32; ++d) acc += (float)qr[d] * (float)kr[d];
      acc = (maskl[m] != 0.f) ? acc : -3.402823466e38f;
      p[m] = acc;
      mx = fmaxf(mx, acc);
    }
    float sum = 0.f;
#pragma unroll
    for (int m = 0; m < 32; ++m) { p[m] = __expf(p[m] - mx); sum += p[m]; }
    float inv = 1.f / sum;
    float o[32];
#pragma unroll
    for (int d = 0; d < 32; ++d) o[d] = 0.f;
    for (int m = 0; m < 32; ++m) {
      float pm = p[m] * inv;
      const _Float16* vr = vhl + (head * 32 + m) * 32;
#pragma unroll
      for (int d = 0; d < 32; ++d) o[d] += pm * (float)vr[d];
    }
    size_t basei = ((size_t)b * N_ + n0 + row) * C_ + head * 32;
#pragma unroll
    for (int d = 0; d < 32; ++d) out[basei + d] = x[basei + d] + o[d];
  }
}

// LN2 + fc1 + GELU + fc2 + residual, dff chunked by 128 (WMMA both GEMMs)
__global__ void mlp_kernel(const _Float16* __restrict__ f1swz, const _Float16* __restrict__ f2swz,
                           const float* __restrict__ fc1_b, const float* __restrict__ fc2_b,
                           const float* __restrict__ g, const float* __restrict__ bb,
                           float* __restrict__ io) {
  extern __shared__ char smem[];
  _Float16* yt = (_Float16*)smem;     // 64 x 264
  _Float16* ht = yt + 64 * 264;       // 64 x 136
  int tid = threadIdx.x, lane = tid & 31, w = tid >> 5;
  size_t row0 = (size_t)blockIdx.x * 64;

  // LN2: wave w handles 8 rows
  for (int i = 0; i < 8; ++i) {
    int r = w * 8 + i;
    const float* xr = io + (row0 + r) * C_ + lane * 8;
    v4f a0 = *(const v4f*)xr;
    v4f a1 = *(const v4f*)(xr + 4);
    float v[8] = {a0[0], a0[1], a0[2], a0[3], a1[0], a1[1], a1[2], a1[3]};
    float s = 0.f, ss = 0.f;
#pragma unroll
    for (int j = 0; j < 8; ++j) { s += v[j]; ss += v[j] * v[j]; }
#pragma unroll
    for (int off = 16; off >= 1; off >>= 1) {
      s += __shfl_xor(s, off, 32);
      ss += __shfl_xor(ss, off, 32);
    }
    float mean = s * (1.f / C_), var = ss * (1.f / C_) - mean * mean;
    float rstd = rsqrtf(var + 1e-5f);
    v8h hv;
#pragma unroll
    for (int j = 0; j < 8; ++j)
      hv[j] = (_Float16)((v[j] - mean) * rstd * g[lane * 8 + j] + bb[lane * 8 + j]);
    *(v8h*)(yt + r * 264 + lane * 8) = hv;
  }
  __syncthreads();

  int rt = w & 3, ch = w >> 2;
  v8f acc[8];
#pragma unroll
  for (int j = 0; j < 8; ++j) { v8f z = {}; acc[j] = z; }

  for (int chunk = 0; chunk < 8; ++chunk) {
    // h = gelu(y @ fc1[:, 128-chunk] + b1): A frags hoisted, each wave 4 of 32 tiles
    v16h ay[8];
#pragma unroll
    for (int kt = 0; kt < 8; ++kt) ay[kt] = load_a_lds(yt, rt * 16, 264, kt * 32, lane);
#pragma unroll
    for (int j = 0; j < 4; ++j) {
      int ctH = ch * 4 + j;
      int ctg = chunk * 8 + ctH;
      v8f a2 = {};
#pragma unroll
      for (int kt = 0; kt < 8; ++kt) {
        a2 = wmma16(ay[kt], load_b_swz(f1swz, kt * 64 + ctg, lane), a2);
      }
      float bias = fc1_b[ctg * 16 + (lane & 15)];
#pragma unroll
      for (int r = 0; r < 8; ++r) {
        float xv = a2[r] + bias;
        a2[r] = 0.5f * xv * (1.0f + erff(xv * 0.7071067811865476f));  // exact GELU
      }
      store_c_f16_lds(ht, rt * 16, 136, ctH * 16, a2, lane);
    }
    __syncthreads();
    if (chunk < 7)
      __builtin_prefetch(f2swz + (size_t)(chunk + 1) * 32768 + tid * 128, 0, 1);
    // out += h @ fc2[128-chunk, :]: A frags hoisted, each wave 8 col-tiles x 4 k-steps
    v16h ah[4];
#pragma unroll
    for (int k2 = 0; k2 < 4; ++k2) ah[k2] = load_a_lds(ht, rt * 16, 136, k2 * 32, lane);
#pragma unroll
    for (int j = 0; j < 8; ++j) {
      int ct = ch * 8 + j;
#pragma unroll
      for (int k2 = 0; k2 < 4; ++k2) {
        acc[j] = wmma16(ah[k2], load_b_swz(f2swz, (chunk * 4 + k2) * 16 + ct, lane), acc[j]);
      }
    }
    __syncthreads();
  }

  // epilogue: + fc2_b + residual (C-layout scatter)
  int ro = (lane >> 4) << 3;
#pragma unroll
  for (int j = 0; j < 8; ++j) {
    int col = (ch * 8 + j) * 16 + (lane & 15);
    float bias = fc2_b[col];
#pragma unroll
    for (int r = 0; r < 8; ++r) {
      size_t idx = (row0 + rt * 16 + ro + r) * C_ + col;
      io[idx] = io[idx] + acc[j][r] + bias;
    }
  }
}

extern "C" void kernel_launch(void* const* d_in, const int* in_sizes, int n_in,
                              void* d_out, int out_size, void* d_ws, size_t ws_size,
                              hipStream_t stream) {
  (void)in_sizes; (void)n_in; (void)out_size; (void)ws_size;
  const float* x    = (const float*)d_in[0];
  const float* rot  = (const float*)d_in[1];
  const float* n1g  = (const float*)d_in[2];
  const float* n1b  = (const float*)d_in[3];
  const float* q_w  = (const float*)d_in[4];
  const float* kv_w = (const float*)d_in[5];
  const float* n2g  = (const float*)d_in[6];
  const float* n2b  = (const float*)d_in[7];
  const float* f1w  = (const float*)d_in[8];
  const float* f1b  = (const float*)d_in[9];
  const float* f2w  = (const float*)d_in[10];
  const float* f2b  = (const float*)d_in[11];
  float* out = (float*)d_out;

  char* ws = (char*)d_ws;
  size_t off = 0;
  auto carve = [&](size_t bytes) -> char* {
    char* p = ws + off;
    off = (off + bytes + 255) & ~(size_t)255;
    return p;
  };
  _Float16* xh   = (_Float16*)carve((size_t)B_ * N_ * C_ * 2);        // 64 MB
  _Float16* qswz = (_Float16*)carve((size_t)C_ * C_ * 2);
  _Float16* f1s  = (_Float16*)carve((size_t)C_ * DFF_ * 2);
  _Float16* f2s  = (_Float16*)carve((size_t)DFF_ * C_ * 2);
  _Float16* kh   = (_Float16*)carve((size_t)B_ * H_ * M_ * DH_ * 2);
  _Float16* vh   = (_Float16*)carve((size_t)B_ * H_ * M_ * DH_ * 2);
  float* pooled  = (float*)carve((size_t)B_ * 32 * C_ * 4);
  float* counts  = (float*)carve((size_t)B_ * 32 * 4);

  // pooled and counts are contiguous (pooled size is 256B-multiple)
  int nzero = B_ * 32 * C_ + B_ * 32;
  zero_kernel<<<(nzero + 255) / 256, 256, 0, stream>>>(pooled, nzero);

  swz_kernel<<<(C_ * C_ + 255) / 256, 256, 0, stream>>>(q_w, qswz, C_, C_);
  swz_kernel<<<(C_ * DFF_ + 255) / 256, 256, 0, stream>>>(f1w, f1s, C_, DFF_);
  swz_kernel<<<(DFF_ * C_ + 255) / 256, 256, 0, stream>>>(f2w, f2s, DFF_, C_);

  ln_hash_kernel<<<dim3(N_ / 256, B_), 256, 51328, stream>>>(
      x, rot, n1g, n1b, xh, pooled, counts);

  kv_kernel<<<B_, 256, 32768, stream>>>(pooled, counts, kv_w, kh, vh);

  q_attn_kernel<<<dim3(N_ / 64, B_), 256, 100480, stream>>>(
      x, xh, qswz, kh, vh, counts, out);

  mlp_kernel<<<(B_ * N_) / 64, 256, 51200, stream>>>(
      f1s, f2s, f1b, f2b, n2g, n2b, out);
}
